// Generator_15814069583897
// MI455X (gfx1250) — matrix operand order repaired and checked by reading, not compiled, not run on hardware
//
#include <hip/hip_runtime.h>
#include <hip/hip_bf16.h>

#define S_ 20
#define B_ 32
#define L_ 128
#define T_ 8
#define H_ 400
#define V_ 20000
#define G_ 3
#define N_ (S_*B_)        // 640 fused rows
#define HP 416            // H padded to multiple of 32 (13 K-slabs)
#define KSLAB (HP/32)     // 13
#define G3H (3*H_)        // 1200
#define G3HP 1216         // gate-GEMM N padded to multiple of 32
#define MT 4              // M tiles per wave (64 rows)
#define NT 2              // N tiles per wave (32 cols)
#define WV 2              // waves per workgroup (share B tiles, split M)

typedef __attribute__((ext_vector_type(16))) __bf16 v16bf;
typedef __attribute__((ext_vector_type(8)))  float  v8f;

struct alignas(16) U4 { unsigned x, y, z, w; };

__device__ __forceinline__ unsigned short f2bf(float f) {
  union { float f; unsigned u; } v; v.f = f;
  unsigned r = v.u + 0x7FFFu + ((v.u >> 16) & 1u);   // round-to-nearest-even
  return (unsigned short)(r >> 16);
}

// ---------------- precompute / build kernels ----------------

// rows_in x H fp32 -> rows_out x HP bf16 (zero pad in K and extra rows)
__global__ void conv_bf16_k(const float* __restrict__ in, unsigned short* __restrict__ out,
                            int rows_in, int rows_out) {
  int i = blockIdx.x * blockDim.x + threadIdx.x;
  if (i >= rows_out * HP) return;
  int n = i / HP, k = i % HP;
  out[i] = (k < H_ && n < rows_in) ? f2bf(in[(size_t)n * H_ + k]) : (unsigned short)0;
}

// h0 = tile(encoded_hidden[0], S) ; writes fp32 + padded bf16
__global__ void init_h0_k(const float* __restrict__ enc_hidden, float* __restrict__ hf,
                          unsigned short* __restrict__ hb) {
  int i = blockIdx.x * blockDim.x + threadIdx.x;
  if (i >= N_ * HP) return;
  int n = i / HP, k = i % HP;
  float v = 0.f;
  if (k < H_) {
    v = enc_hidden[(size_t)(n % B_) * H_ + k];
    hf[(size_t)n * H_ + k] = v;
  }
  hb[(size_t)n * HP + k] = f2bf(v);
}

// decoder input x_t: t==0 -> attr_emb[s]; t>0 -> emb_weight[target[b][s][t-1]]
__global__ void build_x_k(int t, const float* __restrict__ attr, const float* __restrict__ emb,
                          const int* __restrict__ tgt, unsigned short* __restrict__ xb,
                          float* __restrict__ xf) {
  int i = blockIdx.x * blockDim.x + threadIdx.x;
  if (i >= N_ * HP) return;
  int n = i / HP, k = i % HP;
  float v = 0.f;
  if (k < H_) {
    if (t == 0) {
      v = attr[(size_t)(n / B_) * H_ + k];
    } else {
      int b = n % B_, s = n / B_;
      int tok = tgt[(size_t)b * (S_ * T_) + (size_t)s * T_ + (t - 1)];
      v = emb[(size_t)tok * H_ + k];
    }
    xf[(size_t)n * H_ + k] = v;
  }
  xb[(size_t)n * HP + k] = f2bf(v);
}

// ---------------- bf16 WMMA GEMM:  C[M][Nc] = A[M][K] * Bt[Nc][K]^T ----------------
// Workgroup = 2 waves sharing the same 32-col N block (B tiles hit WGP$ twice),
// each wave owns a 64x32 output block (MT=4 x NT=2 register tiles, 8 f32 accumulators).
// Per K-slab: all NT B-frags + MT A-frags loaded up front, then 8 WMMAs — gives the
// scheduler a full slab of loads to overlap with the previous slab's WMMAs.
// A layout (16-bit A 16x32, ISA 7.12.2): lane<16 -> K 0..7 / 16..23; lane>=16 -> +8.
// B layout (dense 32x16): same K order as A: lane<16 -> K 0..7 / 16..23; lane>=16 -> +8, col = lane&15.
__global__ __launch_bounds__(32 * WV)
void wmma_gemm_bf16_k(const unsigned short* __restrict__ A, const unsigned short* __restrict__ Bt,
                      float* __restrict__ C, long long ldc) {
  const int lane  = threadIdx.x;          // 0..31
  const int wv    = threadIdx.y;          // 0..WV-1
  const int l15   = lane & 15;
  const int hi    = lane >> 4;            // 0 or 1
  const int khalf = hi << 3;              // 0 or 8  (A)
  const int kb16  = hi << 4;              // 0 or 16 (B)
  const int mt0   = (blockIdx.y * WV + wv) * (16 * MT);
  const int nt0   = blockIdx.x * (16 * NT);

  const U4* __restrict__ arow[MT];
#pragma unroll
  for (int i = 0; i < MT; ++i)
    arow[i] = (const U4*)(A + (size_t)(mt0 + 16 * i + l15) * HP);
  const U4* __restrict__ brow[NT];
#pragma unroll
  for (int j = 0; j < NT; ++j)
    brow[j] = (const U4*)(Bt + (size_t)(nt0 + 16 * j + l15) * HP);

  v8f acc[MT][NT] = {};
#pragma unroll
  for (int ks = 0; ks < KSLAB; ++ks) {
    const int k0 = ks * 32;
    union FV { U4 q[2]; v16bf v; } b[NT], a[MT];
#pragma unroll
    for (int j = 0; j < NT; ++j) {
      b[j].q[0] = brow[j][(k0 + khalf) >> 3];
      b[j].q[1] = brow[j][((k0 + 16 + khalf) >> 3)];
    }
#pragma unroll
    for (int i = 0; i < MT; ++i) {
      a[i].q[0] = arow[i][(k0 + khalf) >> 3];
      a[i].q[1] = arow[i][((k0 + 16 + khalf) >> 3)];
    }
#pragma unroll
    for (int i = 0; i < MT; ++i)
#pragma unroll
      for (int j = 0; j < NT; ++j)
        acc[i][j] = __builtin_amdgcn_wmma_f32_16x16x32_bf16(false, a[i].v, false, b[j].v,
                                                            (short)0, acc[i][j], false, false);
  }

  const int mlocal = hi << 3;
#pragma unroll
  for (int i = 0; i < MT; ++i) {
    const int mbase = mt0 + 16 * i + mlocal;
#pragma unroll
    for (int j = 0; j < NT; ++j) {
      const int ncol = nt0 + 16 * j + l15;
#pragma unroll
      for (int r = 0; r < 8; ++r)
        C[(size_t)(mbase + r) * ldc + ncol] = acc[i][j][r];
    }
  }
}

// ---------------- GRU elementwise (gate order r,z,n) ----------------
__global__ void gru_ew_k(const float* __restrict__ gi, const float* __restrict__ gh,
                         const float* __restrict__ b_ih, const float* __restrict__ b_hh,
                         const float* __restrict__ h_cur, float* __restrict__ h_new,
                         unsigned short* __restrict__ h_new_b) {
  int i = blockIdx.x * blockDim.x + threadIdx.x;
  if (i >= N_ * HP) return;
  int n = i / HP, k = i % HP;
  if (k >= H_) { h_new_b[(size_t)n * HP + k] = 0; return; }
  const float* gin = gi + (size_t)n * G3HP;
  const float* ghn = gh + (size_t)n * G3HP;
  float ir = gin[k]          + b_ih[k];
  float iz = gin[H_ + k]     + b_ih[H_ + k];
  float in_ = gin[2*H_ + k]  + b_ih[2*H_ + k];
  float hr = ghn[k]          + b_hh[k];
  float hz = ghn[H_ + k]     + b_hh[H_ + k];
  float hn = ghn[2*H_ + k]   + b_hh[2*H_ + k];
  float r = 1.f / (1.f + expf(-(ir + hr)));
  float z = 1.f / (1.f + expf(-(iz + hz)));
  float nn = tanhf(in_ + r * hn);
  float hv = (1.f - z) * nn + z * h_cur[(size_t)n * H_ + k];
  h_new[(size_t)n * H_ + k] = hv;
  h_new_b[(size_t)n * HP + k] = f2bf(hv);
}

// ---------------- attention: scores -> masked softmax(L) -> prob, context ----------------
__global__ void attn_k(const float* __restrict__ h_new, const float* __restrict__ enc_out,
                       const int* __restrict__ lens, float* __restrict__ prob,
                       float* __restrict__ ctx) {
  const int n = blockIdx.x, tid = threadIdx.x;   // block = 128 = L_
  const int b = n % B_;
  __shared__ float hl[H_];
  __shared__ float pl[L_];
  __shared__ float red[L_];
  for (int h = tid; h < H_; h += L_) hl[h] = h_new[(size_t)n * H_ + h];
  __syncthreads();
  const int len = lens[b];
  const float* encb = enc_out + (size_t)b * L_ * H_;
  float s = -INFINITY;
  if (tid < len) {
    const float* e = encb + (size_t)tid * H_;
    s = 0.f;
    for (int h = 0; h < H_; ++h) s += hl[h] * e[h];
  }
  red[tid] = s; __syncthreads();
  for (int off = 64; off > 0; off >>= 1) { if (tid < off) red[tid] = fmaxf(red[tid], red[tid + off]); __syncthreads(); }
  float mx = red[0]; __syncthreads();
  float ev = (tid < len) ? expf(s - mx) : 0.f;
  red[tid] = ev; __syncthreads();
  for (int off = 64; off > 0; off >>= 1) { if (tid < off) red[tid] += red[tid + off]; __syncthreads(); }
  float p = ev / red[0];
  pl[tid] = p;
  prob[(size_t)n * L_ + tid] = p;
  __syncthreads();
  for (int h = tid; h < H_; h += L_) {
    float acc = 0.f;
    for (int l = 0; l < L_; ++l) acc += pl[l] * encb[(size_t)l * H_ + h];
    ctx[(size_t)n * H_ + h] = acc;
  }
}

// ---------------- pointer-generator switch ----------------
__global__ void switch_k(const float* __restrict__ h_new, const float* __restrict__ ctx,
                         const float* __restrict__ xf, const float* __restrict__ Wr,
                         const float* __restrict__ br, float* __restrict__ sw) {
  const int n = blockIdx.x, tid = threadIdx.x;   // block = 128
  __shared__ float red[128];
  float acc = 0.f;
  for (int j = tid; j < G3H; j += 128) {
    float v = (j < H_) ? h_new[(size_t)n * H_ + j]
            : (j < 2*H_) ? ctx[(size_t)n * H_ + (j - H_)]
                         : xf[(size_t)n * H_ + (j - 2*H_)];
    acc += Wr[j] * v;
  }
  red[tid] = acc; __syncthreads();
  for (int off = 64; off > 0; off >>= 1) { if (tid < off) red[tid] += red[tid + off]; __syncthreads(); }
  if (tid == 0) sw[n] = 1.f / (1.f + expf(-(red[0] + br[0])));
}

// ---------------- gate head from step-0 context ----------------
__global__ void gate_k(const float* __restrict__ ctx, const float* __restrict__ Wg,
                       const float* __restrict__ bg, float* __restrict__ outg) {
  const int idx = blockIdx.x;                 // N_*G_ blocks
  const int n = idx / G_, g = idx % G_, tid = threadIdx.x; // block = 128
  __shared__ float red[128];
  float acc = 0.f;
  for (int h = tid; h < H_; h += 128) acc += ctx[(size_t)n * H_ + h] * Wg[(size_t)g * H_ + h];
  red[tid] = acc; __syncthreads();
  for (int off = 64; off > 0; off >>= 1) { if (tid < off) red[tid] += red[tid + off]; __syncthreads(); }
  if (tid == 0) outg[idx] = red[0] + bg[g];
}

// ---------------- in-place softmax over V and pointer-gen scaling ----------------
__global__ void softmax_final_k(float* __restrict__ base, const float* __restrict__ sw, long long ldc) {
  const int n = blockIdx.x, tid = threadIdx.x;  // block = 256
  float* p = base + (size_t)n * ldc;
  __shared__ float red[256];
  float mx = -INFINITY;
  for (int v = tid; v < V_; v += 256) mx = fmaxf(mx, p[v]);
  red[tid] = mx; __syncthreads();
  for (int off = 128; off > 0; off >>= 1) { if (tid < off) red[tid] = fmaxf(red[tid], red[tid + off]); __syncthreads(); }
  mx = red[0]; __syncthreads();
  float s = 0.f;
  for (int v = tid; v < V_; v += 256) { float e = expf(p[v] - mx); p[v] = e; s += e; }
  red[tid] = s; __syncthreads();
  for (int off = 128; off > 0; off >>= 1) { if (tid < off) red[tid] += red[tid + off]; __syncthreads(); }
  float scale = sw[n] / red[0];
  for (int v = tid; v < V_; v += 256) p[v] *= scale;
}

// ---------------- sparse scatter of (1-switch)*prob onto story tokens ----------------
__global__ void scatter_k(float* __restrict__ base, long long ldc, const float* __restrict__ prob,
                          const float* __restrict__ sw, const int* __restrict__ story,
                          const int* __restrict__ lens) {
  const int n = blockIdx.x, l = threadIdx.x;    // block = 128
  const int b = n % B_;
  if (l < lens[b]) {
    float c = (1.f - sw[n]) * prob[(size_t)n * L_ + l];
    atomicAdd(base + (size_t)n * ldc + story[(size_t)b * L_ + l], c);
  }
}

extern "C" void kernel_launch(void* const* d_in, const int* in_sizes, int n_in,
                              void* d_out, int out_size, void* d_ws, size_t ws_size,
                              hipStream_t stream) {
  (void)in_sizes; (void)n_in; (void)out_size; (void)ws_size;
  const float* enc_hidden = (const float*)d_in[0];
  const float* enc_out    = (const float*)d_in[1];
  const int*   enc_lens   = (const int*)d_in[2];
  const int*   story      = (const int*)d_in[3];
  const int*   tgt        = (const int*)d_in[4];
  const float* emb        = (const float*)d_in[5];
  const float* attr       = (const float*)d_in[6];
  const float* w_ih       = (const float*)d_in[7];
  const float* w_hh       = (const float*)d_in[8];
  const float* b_ih       = (const float*)d_in[9];
  const float* b_hh       = (const float*)d_in[10];
  const float* Wr         = (const float*)d_in[11];
  const float* br         = (const float*)d_in[12];
  const float* Wg         = (const float*)d_in[13];
  const float* bg         = (const float*)d_in[14];
  float* out      = (float*)d_out;
  float* out_gate = out + (size_t)S_ * B_ * T_ * V_;

  char* ws = (char*)d_ws;
  size_t cur = 0;
  auto alloc = [&](size_t bytes) -> char* {
    size_t off = cur; cur = (cur + bytes + 255) & ~(size_t)255; return ws + off;
  };
  unsigned short* emb_b = (unsigned short*)alloc((size_t)V_ * HP * 2);
  unsigned short* wih_b = (unsigned short*)alloc((size_t)G3HP * HP * 2);
  unsigned short* whh_b = (unsigned short*)alloc((size_t)G3HP * HP * 2);
  unsigned short* x_b   = (unsigned short*)alloc((size_t)N_ * HP * 2);
  float* x_f  = (float*)alloc((size_t)N_ * H_ * 4);
  float* hA_f = (float*)alloc((size_t)N_ * H_ * 4);
  unsigned short* hA_b = (unsigned short*)alloc((size_t)N_ * HP * 2);
  float* hB_f = (float*)alloc((size_t)N_ * H_ * 4);
  unsigned short* hB_b = (unsigned short*)alloc((size_t)N_ * HP * 2);
  float* gi   = (float*)alloc((size_t)N_ * G3HP * 4);
  float* gh   = (float*)alloc((size_t)N_ * G3HP * 4);
  float* prob = (float*)alloc((size_t)N_ * L_ * 4);
  float* ctx  = (float*)alloc((size_t)N_ * H_ * 4);
  float* sw   = (float*)alloc((size_t)N_ * 4);

  const long long ldcV = (long long)T_ * V_;

  // precompute bf16 operands + h0
  conv_bf16_k<<<((size_t)V_ * HP + 255) / 256, 256, 0, stream>>>(emb, emb_b, V_, V_);
  conv_bf16_k<<<((size_t)G3HP * HP + 255) / 256, 256, 0, stream>>>(w_ih, wih_b, G3H, G3HP);
  conv_bf16_k<<<((size_t)G3HP * HP + 255) / 256, 256, 0, stream>>>(w_hh, whh_b, G3H, G3HP);
  init_h0_k<<<(N_ * HP + 255) / 256, 256, 0, stream>>>(enc_hidden, hA_f, hA_b);

  float* hc_f = hA_f; unsigned short* hc_b = hA_b;
  float* hn_f = hB_f; unsigned short* hn_b = hB_b;

  const dim3 blkGemm(32, WV);
  const dim3 gridGate(G3HP / (16 * NT), N_ / (16 * MT * WV));   // (38, 5)
  const dim3 gridVocab(V_ / (16 * NT),  N_ / (16 * MT * WV));   // (625, 5)

  for (int t = 0; t < T_; ++t) {
    build_x_k<<<(N_ * HP + 255) / 256, 256, 0, stream>>>(t, attr, emb, tgt, x_b, x_f);
    // gi = x @ W_ih^T ; gh = h @ W_hh^T
    wmma_gemm_bf16_k<<<gridGate, blkGemm, 0, stream>>>(x_b,  wih_b, gi, (long long)G3HP);
    wmma_gemm_bf16_k<<<gridGate, blkGemm, 0, stream>>>(hc_b, whh_b, gh, (long long)G3HP);
    gru_ew_k<<<(N_ * HP + 255) / 256, 256, 0, stream>>>(gi, gh, b_ih, b_hh, hc_f, hn_f, hn_b);
    attn_k<<<N_, L_, 0, stream>>>(hn_f, enc_out, enc_lens, prob, ctx);
    if (t == 0)
      gate_k<<<N_ * G_, 128, 0, stream>>>(ctx, Wg, bg, out_gate);
    switch_k<<<N_, 128, 0, stream>>>(hn_f, ctx, x_f, Wr, br, sw);
    // logits = h_new @ emb^T, written straight into d_out (row stride T*V)
    wmma_gemm_bf16_k<<<gridVocab, blkGemm, 0, stream>>>(hn_b, emb_b, out + (size_t)t * V_, ldcV);
    softmax_final_k<<<N_, 256, 0, stream>>>(out + (size_t)t * V_, sw, ldcV);
    scatter_k<<<N_, L_, 0, stream>>>(out + (size_t)t * V_, ldcV, prob, sw, story, enc_lens);
    // ping-pong hidden state
    float* tf = hc_f; hc_f = hn_f; hn_f = tf;
    unsigned short* tb = hc_b; hc_b = hn_b; hn_b = tb;
  }
}
